// Q_Attention_63058709840189
// MI455X (gfx1250) — 16-bit and 8-bit matrix operand order repaired and checked by reading, compiled, not run on hardware
//
#include <hip/hip_runtime.h>
#include <cstdint>
#include <cstddef>


#define Bc 64
#define Nc 198
#define Cc 768
#define Hc 12
#define Dc 64
#define NP 208
#define KP 224

typedef __attribute__((ext_vector_type(8)))  int       v8i;
typedef __attribute__((ext_vector_type(16))) int       v16i;
typedef __attribute__((ext_vector_type(8)))  float     v8f;
typedef __attribute__((ext_vector_type(16))) _Float16  v16h;

typedef union { v16h h; v8i i; } f16frag;

__device__ __forceinline__ v8i load_a_b8(const uint8_t* p, int lda, int row0,
                                         int k0, int lane) {
  int r  = row0 + (lane & 15);
  int kb = k0 + ((lane >> 4) << 3);
  const uint8_t* base = p + (size_t)r * lda + kb;
  v8i a;
  a[0] = *(const int*)(base +  0);
  a[1] = *(const int*)(base +  4);
  a[2] = *(const int*)(base + 16);
  a[3] = *(const int*)(base + 20);
  a[4] = *(const int*)(base + 32);
  a[5] = *(const int*)(base + 36);
  a[6] = *(const int*)(base + 48);
  a[7] = *(const int*)(base + 52);
  return a;
}

__device__ __forceinline__ v8i load_b_b8(const uint8_t* p, int ldb, int n0,
                                         int k0, int lane) {
  int nn = n0 + (lane & 15);
  int kb = k0 + ((lane >> 4) << 3);
  const uint8_t* base = p + (size_t)nn * ldb + kb;
  v8i b;
  b[0] = *(const int*)(base +  0);
  b[1] = *(const int*)(base +  4);
  b[2] = *(const int*)(base + 16);
  b[3] = *(const int*)(base + 20);
  b[4] = *(const int*)(base + 32);
  b[5] = *(const int*)(base + 36);
  b[6] = *(const int*)(base + 48);
  b[7] = *(const int*)(base + 52);
  return b;
}

__device__ __forceinline__ v16i load_a_fp8_128_packed(const uint8_t* p, int K,
                                                      int m0, int k0,
                                                      int lane) {
  size_t base =
      ((size_t)(m0 >> 4) * (K >> 6) + (k0 >> 6)) * 1024 + (size_t)lane * 32;
  v8i lo = *(const v8i*)(p + base);
  v8i hi = *(const v8i*)(p + base + 1024);
  v16i r;
#pragma unroll
  for (int j = 0; j < 8; ++j) { r[j] = lo[j]; r[8 + j] = hi[j]; }
  return r;
}

__device__ __forceinline__ v16i load_b_fp8_128(const uint8_t* p, int ldb,
                                               int n0, int k0, int lane) {
  v8i lo = load_b_b8(p, ldb, n0, k0, lane);
  v8i hi = load_b_b8(p, ldb, n0, k0 + 64, lane);
  v16i r;
#pragma unroll
  for (int j = 0; j < 8; ++j) { r[j] = lo[j]; r[8 + j] = hi[j]; }
  return r;
}

__device__ __forceinline__ v16h load_a_f16(const _Float16* p, int lda, int row0,
                                           int k0, int lane) {
  int r  = row0 + (lane & 15);
  int kb = k0 + ((lane >> 4) << 4);
  const _Float16* base = p + (size_t)r * lda;
  f16frag f;
  f.i[0] = *(const int*)(base + kb +  0);
  f.i[1] = *(const int*)(base + kb +  2);
  f.i[2] = *(const int*)(base + kb +  4);
  f.i[3] = *(const int*)(base + kb +  6);
  f.i[4] = *(const int*)(base + kb +  8);
  f.i[5] = *(const int*)(base + kb + 10);
  f.i[6] = *(const int*)(base + kb + 12);
  f.i[7] = *(const int*)(base + kb + 14);
  return f.h;
}

__device__ __forceinline__ v16h load_b_f16(const _Float16* p, int ldb, int n0,
                                           int k0, int lane) {
  int nn = n0 + (lane & 15);
  int kb = k0 + ((lane >> 4) << 4);
  const unsigned short* ps = (const unsigned short*)p;
  f16frag f;
#pragma unroll
  for (int j = 0; j < 8; ++j) {
    unsigned lo = ps[(size_t)(kb + 2 * j)     * ldb + nn];
    unsigned hi = ps[(size_t)(kb + 2 * j + 1) * ldb + nn];
    f.i[j] = (int)(lo | (hi << 16));
  }
  return f.h;
}

__device__ __forceinline__ uint8_t fp8sign(float v) {
  return (v > 0.f) ? (uint8_t)0x38 : ((v < 0.f) ? (uint8_t)0xB8 : (uint8_t)0);
}
__device__ __forceinline__ float fsign(float v) {
  return (v > 0.f) ? 1.f : ((v < 0.f) ? -1.f : 0.f);
}

__global__ void abs_sum_kernel(const float* __restrict__ w, int n,
                               float* __restrict__ out) {
  __shared__ float red[256];
  float s = 0.f;
  for (int i = blockIdx.x * blockDim.x + threadIdx.x; i < n;
       i += gridDim.x * blockDim.x)
    s += fabsf(w[i]);
  red[threadIdx.x] = s;
  __syncthreads();
  for (int off = 128; off > 0; off >>= 1) {
    if (threadIdx.x < (unsigned)off) red[threadIdx.x] += red[threadIdx.x + off];
    __syncthreads();
  }
  if (threadIdx.x == 0) atomicAdd(out, red[0]);
}

__global__ void sign_fp8_kernel(const float* __restrict__ x,
                                uint8_t* __restrict__ out, int n) {
  int i = blockIdx.x * blockDim.x + threadIdx.x;
  if (i < n) out[i] = fp8sign(x[i]);
}

__global__ void sign_fp8_pack_a_kernel(const float* __restrict__ x,
                                       uint8_t* __restrict__ out, int K) {
  int m = blockIdx.y;
  int k = blockIdx.x * blockDim.x + threadIdx.x;
  float v = x[(size_t)m * K + k];
  int tile = (m >> 4) * (K >> 6) + (k >> 6);
  int rm = m & 15, kk = k & 63;
  int lane = rm + (((kk >> 3) & 1) << 4);
  int pos = ((kk >> 4) << 3) + (kk & 7);
  out[(size_t)tile * 1024 + lane * 32 + pos] = fp8sign(v);
}

__global__ void gemm_fp8_kernel(const uint8_t* __restrict__ A,
                                const uint8_t* __restrict__ Bm,
                                float* __restrict__ C, int Nn, int K,
                                const float* __restrict__ act_clip,
                                const float* __restrict__ wsum, float wnum_inv,
                                const float* __restrict__ bias) {
  __shared__ uint8_t Bsh[64 * 128];
  const int lane = threadIdx.x & 31;
  const int wave = threadIdx.x >> 5;
  const int m0 = blockIdx.x * 128 + (wave >> 1) * 32;
  const int nblk = blockIdx.y * 64;
  const int n0 = (wave & 1) * 32;

  const unsigned ldsbase = (unsigned)(size_t)(void*)Bsh;
  const int rowc = threadIdx.x >> 3;
  const int segc = (threadIdx.x & 7) << 4;

  v8f acc00 = {}, acc01 = {}, acc10 = {}, acc11 = {};
  for (int k0 = 0; k0 < K; k0 += 128) {
    {
      unsigned long long g0 =
          (unsigned long long)(Bm + (size_t)(nblk + rowc) * K + k0 + segc);
      unsigned l0 = ldsbase + rowc * 128 + segc;
      asm volatile("global_load_async_to_lds_b128 %0, %1, off"
                   :: "v"(l0), "v"(g0) : "memory");
      unsigned long long g1 =
          (unsigned long long)(Bm + (size_t)(nblk + rowc + 32) * K + k0 + segc);
      unsigned l1 = ldsbase + (rowc + 32) * 128 + segc;
      asm volatile("global_load_async_to_lds_b128 %0, %1, off"
                   :: "v"(l1), "v"(g1) : "memory");
    }
    __builtin_prefetch(
        A + ((size_t)(m0 >> 4) * (K >> 6) + ((k0 + 128) >> 6)) * 1024, 0, 1);
    v16i a0 = load_a_fp8_128_packed(A, K, m0, k0, lane);
    v16i a1 = load_a_fp8_128_packed(A, K, m0 + 16, k0, lane);
    asm volatile("s_wait_asynccnt 0" ::: "memory");
    __syncthreads();
    v16i b0 = load_b_fp8_128(Bsh, 128, n0, 0, lane);
    v16i b1 = load_b_fp8_128(Bsh, 128, n0 + 16, 0, lane);
    acc00 = __builtin_amdgcn_wmma_f32_16x16x128_fp8_fp8(a0, b0, (short)0,
                                                        acc00, false, false);
    acc01 = __builtin_amdgcn_wmma_f32_16x16x128_fp8_fp8(a0, b1, (short)0,
                                                        acc01, false, false);
    acc10 = __builtin_amdgcn_wmma_f32_16x16x128_fp8_fp8(a1, b0, (short)0,
                                                        acc10, false, false);
    acc11 = __builtin_amdgcn_wmma_f32_16x16x128_fp8_fp8(a1, b1, (short)0,
                                                        acc11, false, false);
    __syncthreads();
  }

  const float scale = (*act_clip) * ((*wsum) * wnum_inv);
  const int colb = lane & 15;
  const int rowoff = (lane >> 4) * 8;
  const int ng = nblk + n0;
#pragma unroll
  for (int i = 0; i < 8; ++i) {
    int r = m0 + i + rowoff;
    float b0v = bias ? bias[ng + colb] : 0.f;
    float b1v = bias ? bias[ng + 16 + colb] : 0.f;
    C[(size_t)r * Nn + ng + colb]             = acc00[i] * scale + b0v;
    C[(size_t)r * Nn + ng + 16 + colb]        = acc01[i] * scale + b1v;
    C[(size_t)(r + 16) * Nn + ng + colb]      = acc10[i] * scale + b0v;
    C[(size_t)(r + 16) * Nn + ng + 16 + colb] = acc11[i] * scale + b1v;
  }
}

__global__ void attention_kernel(const float* __restrict__ qkvbuf,
                                 float* __restrict__ outbuf,
                                 const float* __restrict__ move_q_b,
                                 const float* __restrict__ move_k_b,
                                 const float* __restrict__ move_v_b,
                                 const float* __restrict__ move_att_b,
                                 const float* __restrict__ cq,
                                 const float* __restrict__ ck,
                                 const float* __restrict__ cv1,
                                 const float* __restrict__ cv2,
                                 const float* __restrict__ cv3,
                                 const float* __restrict__ ca1,
                                 const float* __restrict__ ca2,
                                 const float* __restrict__ ca3) {
  const int b = blockIdx.x;
  const int h = blockIdx.y;

  extern __shared__ char smem[];
  uint8_t*  q8   = (uint8_t*)smem;
  uint8_t*  k8   = q8 + NP * Dc;
  _Float16* vh   = (_Float16*)(k8 + NP * Dc);
  float*    Sall = (float*)((char*)vh + (size_t)KP * Dc * 2);
  _Float16* Aall = (_Float16*)((char*)Sall + (size_t)8 * 16 * KP * 4);

  const int tid = threadIdx.x;
  const int lane = tid & 31;
  const int wave = tid >> 5;

  const float va[3] = {*cv1, *cv2, *cv3};
  const float aa[3] = {*ca1, *ca2, *ca3};
  const float* base = qkvbuf + (size_t)b * Nc * (3 * Cc);

  for (int idx = tid; idx < NP * Dc; idx += blockDim.x) {
    int n = idx >> 6, d = idx & 63;
    uint8_t qs = 0, ks = 0;
    if (n < Nc) {
      float qv = base[(size_t)n * (3 * Cc) + h * Dc + d] + move_q_b[h * Dc + d];
      float kv =
          base[(size_t)n * (3 * Cc) + Cc + h * Dc + d] + move_k_b[h * Dc + d];
      qs = fp8sign(qv);
      ks = fp8sign(kv);
    }
    q8[idx] = qs;
    k8[idx] = ks;
  }
  for (int idx = tid; idx < KP * Dc; idx += blockDim.x) {
    int m = idx >> 6, d = idx & 63;
    float o = 0.f;
    if (m < Nc) {
      float r = base[(size_t)m * (3 * Cc) + 2 * Cc + h * Dc + d] +
                move_v_b[h * Dc + d];
#pragma unroll
      for (int j = 0; j < 3; ++j) {
        float bq = va[j] * fsign(r);
        o += bq;
        r -= bq;
      }
    }
    vh[idx] = (_Float16)o;
  }
  __syncthreads();

  const float sscale = (*cq) * (*ck) * 0.125f;
  const int rowoff = (lane >> 4) * 8;
  const int colb = lane & 15;

  for (int strip = wave; strip < 13; strip += 8) {
    float*    S  = Sall + (size_t)wave * 16 * KP;
    _Float16* Ah = Aall + (size_t)wave * 16 * KP;
    const int row0 = strip * 16;

    v8i afrag = load_a_b8(q8, Dc, row0, 0, lane);
#pragma unroll
    for (int jt = 0; jt < 13; ++jt) {
      v8i bfrag = load_b_b8(k8, Dc, jt * 16, 0, lane);
      v8f acc = {};
      acc = __builtin_amdgcn_wmma_f32_16x16x64_fp8_fp8(afrag, bfrag, (short)0,
                                                       acc, false, false);
#pragma unroll
      for (int i = 0; i < 8; ++i)
        S[(i + rowoff) * KP + jt * 16 + colb] = acc[i] * sscale;
    }

    {
      const int rr = lane >> 1;
      const int part = lane & 1;
      const int n = row0 + rr;
      float* Srow = S + rr * KP;
      _Float16* Arow = Ah + rr * KP;
      if (n < Nc) {
        float mx = -3.4e38f;
        for (int m = part; m < Nc; m += 2) mx = fmaxf(mx, Srow[m]);
        mx = fmaxf(mx, __shfl_xor(mx, 1));
        float sum = 0.f;
        for (int m = part; m < Nc; m += 2) sum += __expf(Srow[m] - mx);
        sum += __shfl_xor(sum, 1);
        float inv = 1.f / sum;
        const float* mab = move_att_b + ((size_t)h * Nc + n) * Nc;
        for (int m = part; m < KP; m += 2) {
          _Float16 hv = (_Float16)0.f;
          if (m < Nc) {
            float p = __expf(Srow[m] - mx) * inv + mab[m];
            float o = 0.f, r = p;
#pragma unroll
            for (int j = 0; j < 3; ++j) {
              float qh = fminf(fmaxf(rintf(r / aa[j]), 0.f), 1.f);
              float bq = aa[j] * qh;
              o += bq;
              r -= bq;
            }
            hv = (_Float16)o;
          }
          Arow[m] = hv;
        }
      } else {
        for (int m = part; m < KP; m += 2) Arow[m] = (_Float16)0.f;
      }
    }

#pragma unroll
    for (int dt = 0; dt < 4; ++dt) {
      v8f acc = {};
#pragma unroll
      for (int kt = 0; kt < 7; ++kt) {
        v16h af = load_a_f16(Ah, KP, 0, kt * 32, lane);
        v16h bf = load_b_f16(vh, Dc, dt * 16, kt * 32, lane);
        acc = __builtin_amdgcn_wmma_f32_16x16x32_f16(false, af, false, bf,
                                                     (short)0, acc, false,
                                                     false);
      }
#pragma unroll
      for (int i = 0; i < 8; ++i) {
        int no = row0 + i + rowoff;
        if (no < Nc)
          outbuf[((size_t)b * Nc + no) * Cc + h * Dc + dt * 16 + colb] = acc[i];
      }
    }
  }
}

extern "C" void kernel_launch(void* const* d_in, const int* in_sizes, int n_in,
                              void* d_out, int out_size, void* d_ws,
                              size_t ws_size, hipStream_t stream) {
  (void)in_sizes; (void)n_in; (void)out_size; (void)ws_size;
  const float* x        = (const float*)d_in[0];
  const float* qkv_w    = (const float*)d_in[1];
  const float* proj_w   = (const float*)d_in[2];
  const float* proj_b   = (const float*)d_in[3];
  const float* move_q_b = (const float*)d_in[4];
  const float* move_k_b = (const float*)d_in[5];
  const float* move_v_b = (const float*)d_in[6];
  const float* move_att = (const float*)d_in[7];
  const float* clip_in_qkv  = (const float*)d_in[8];
  const float* clip_query   = (const float*)d_in[9];
  const float* clip_key     = (const float*)d_in[10];
  const float* clip_v1      = (const float*)d_in[11];
  const float* clip_v2      = (const float*)d_in[12];
  const float* clip_v3      = (const float*)d_in[13];
  const float* clip_a1      = (const float*)d_in[14];
  const float* clip_a2      = (const float*)d_in[15];
  const float* clip_a3      = (const float*)d_in[16];
  const float* clip_in_proj = (const float*)d_in[17];
  float* out = (float*)d_out;

  const int M = Bc * Nc;
  const size_t sz_x   = (size_t)M * Cc;
  const size_t sz_wq  = (size_t)3 * Cc * Cc;
  const size_t sz_wp  = (size_t)Cc * Cc;
  const size_t sz_qkv = (size_t)M * 3 * Cc;

  char* w = (char*)d_ws;
  float*   scales  = (float*)w;                   size_t off = 256;
  uint8_t* xq8     = (uint8_t*)(w + off);         off += sz_x;
  uint8_t* wq8     = (uint8_t*)(w + off);         off += sz_wq;
  uint8_t* wp8     = (uint8_t*)(w + off);         off += sz_wp;
  float*   qkvbuf  = (float*)(w + off);           off += sz_qkv * 4;
  float*   attnout = (float*)(w + off);           off += sz_x * 4;
  uint8_t* po8     = (uint8_t*)(w + off);         off += sz_x;

  hipMemsetAsync(scales, 0, 2 * sizeof(float), stream);

  abs_sum_kernel<<<256, 256, 0, stream>>>(qkv_w, (int)sz_wq, &scales[0]);
  abs_sum_kernel<<<256, 256, 0, stream>>>(proj_w, (int)sz_wp, &scales[1]);

  sign_fp8_pack_a_kernel<<<dim3(Cc / 256, M), 256, 0, stream>>>(x, xq8, Cc);
  sign_fp8_kernel<<<(int)((sz_wq + 255) / 256), 256, 0, stream>>>(qkv_w, wq8,
                                                                  (int)sz_wq);
  sign_fp8_kernel<<<(int)((sz_wp + 255) / 256), 256, 0, stream>>>(proj_w, wp8,
                                                                  (int)sz_wp);

  gemm_fp8_kernel<<<dim3(M / 128, (3 * Cc) / 64), 256, 0, stream>>>(
      xq8, wq8, qkvbuf, 3 * Cc, Cc, clip_in_qkv, &scales[0],
      1.f / (float)sz_wq, nullptr);

  size_t smem = (size_t)NP * Dc + (size_t)NP * Dc + (size_t)KP * Dc * 2 +
                (size_t)8 * 16 * KP * 4 + (size_t)8 * 16 * KP * 2;
  attention_kernel<<<dim3(Bc, Hc), 256, smem, stream>>>(
      qkvbuf, attnout, move_q_b, move_k_b, move_v_b, move_att, clip_query,
      clip_key, clip_v1, clip_v2, clip_v3, clip_a1, clip_a2, clip_a3);

  sign_fp8_pack_a_kernel<<<dim3(Cc / 256, M), 256, 0, stream>>>(attnout, po8,
                                                                Cc);
  gemm_fp8_kernel<<<dim3(M / 128, Cc / 64), 256, 0, stream>>>(
      po8, wp8, out, Cc, Cc, clip_in_proj, &scales[1], 1.f / (float)sz_wp,
      proj_b);
}
